// MatrixFFT_91328184582437
// MI455X (gfx1250) — matrix operand order repaired and checked by reading, not compiled, not run on hardware
//
#include <hip/hip_runtime.h>
#include <hip/hip_bf16.h>

typedef __attribute__((ext_vector_type(16))) _Float16 v16h;
typedef __attribute__((ext_vector_type(8)))  float    v8f;

#define N_FFT 4096
#define KT_TILES (N_FFT / 32)   // 128 K-tiles of 32
#define NT_TILES (N_FFT / 16)   // 256 N-tiles of 16
#define W_TILE_BYTES 2048       // Wr tile (1024 B) + Wi tile (1024 B)
#define X_TILE_BYTES 1024       // one 16x32 f16 A-fragment tile

// ---------------------------------------------------------------------------
// Kernel 1: generate causal DFT matrix W (real & imag) directly in WMMA
// B-fragment layout. Tile (nt, kt) covers B[k = kt*32 .. +31][n = nt*16 .. +15]
// with B[k][n] = W[n][k] (masked to n >= k), stored per-lane as 8 dwords:
//   lanes 0-15 : N = lane,    K = kbase+0..15  (dword v -> K = 2v, 2v+1)
//   lanes 16-31: N = lane-16, K = kbase+16..31
// ---------------------------------------------------------------------------
__global__ void __launch_bounds__(256) gen_w_kernel(unsigned char* __restrict__ wsW) {
  const int lane = threadIdx.x & 31;
  const int wave = threadIdx.x >> 5;
  const int t  = blockIdx.x * 8 + wave;        // tile id = nt*KT_TILES + kt
  const int nt = t / KT_TILES;
  const int kt = t - nt * KT_TILES;
  const int n  = nt * 16 + (lane & 15);
  const int kbase = kt * 32 + ((lane & 16) ? 16 : 0);

  const float w0    = -6.28318530717958647692f / (float)N_FFT;
  const float scale = 0.015625f;               // 1/sqrt(4096)

  v16h hr, hi;
  #pragma unroll
  for (int v = 0; v < 8; ++v) {
    #pragma unroll
    for (int e = 0; e < 2; ++e) {
      const int k = kbase + 2 * v + e;
      float cr = 0.0f, ci = 0.0f;
      if (n >= k) {                            // causal (lower-triangular) mask
        const unsigned mm = ((unsigned)n * (unsigned)k) & (unsigned)(N_FFT - 1);
        const float th = w0 * (float)mm;
        float s, c;
        __sincosf(th, &s, &c);
        cr = c * scale;
        ci = s * scale;
      }
      hr[2 * v + e] = (_Float16)cr;
      hi[2 * v + e] = (_Float16)ci;
    }
  }
  unsigned char* base = wsW + (size_t)t * W_TILE_BYTES + (size_t)lane * 32;
  *(v16h*)(base)        = hr;                  // Wr fragment
  *(v16h*)(base + 1024) = hi;                  // Wi fragment
}

// ---------------------------------------------------------------------------
// Kernel 2: convert x (f32) into f16 WMMA A-fragments. Tile (mt, kt) covers
// A[m = mt*16 .. +15][k = kt*32 .. +31], per-lane 8 dwords:
//   lanes 0-15 : M = lane,    dwords 0-3 -> K 0..7,   dwords 4-7 -> K 8..15
//   lanes 16-31: M = lane-16, dwords 0-3 -> K 16..23, dwords 4-7 -> K 24..31
// ---------------------------------------------------------------------------
__global__ void __launch_bounds__(256) gen_x_kernel(const float* __restrict__ x,
                                                    unsigned char* __restrict__ wsX) {
  const int lane = threadIdx.x & 31;
  const int wave = threadIdx.x >> 5;
  const int t  = blockIdx.x * 8 + wave;        // tile id = mt*KT_TILES + kt
  const int mt = t / KT_TILES;
  const int kt = t - mt * KT_TILES;
  const int m  = mt * 16 + (lane & 15);
  const int kb = kt * 32 + ((lane & 16) ? 16 : 0);

  const float* row = x + (size_t)m * N_FFT;
  v16h hx;
  #pragma unroll
  for (int v = 0; v < 4; ++v) {
    const float2 f = *(const float2*)(row + kb + 2 * v);
    hx[2 * v]     = (_Float16)f.x;
    hx[2 * v + 1] = (_Float16)f.y;
  }
  #pragma unroll
  for (int v = 0; v < 4; ++v) {
    const float2 f = *(const float2*)(row + kb + 8 + 2 * v);
    hx[8 + 2 * v]     = (_Float16)f.x;
    hx[8 + 2 * v + 1] = (_Float16)f.y;
  }
  *(v16h*)(wsX + (size_t)t * X_TILE_BYTES + (size_t)lane * 32) = hx;
}

// ---------------------------------------------------------------------------
// Kernel 3: fused real+imag causal-DFT GEMM.
// Block = 8 waves (4 along M x 2 along N) covering 128(M) x 128(N).
// Each wave owns 2(M) x 4(N) 16x16 subtiles with real+imag f32 accumulators
// (16 x v8f = 128 VGPRs). Inner K-step (K=32): 10 fragment loads (20 x b128),
// 16 x v_wmma_f32_16x16x32_f16  ->  ~26 FLOP per fragment byte.
// Whole working set (96 MB) is L2-resident on MI455X (192 MB L2).
// ---------------------------------------------------------------------------
__global__ void __launch_bounds__(256) causal_dft_wmma_kernel(
    const unsigned char* __restrict__ wsW,
    const unsigned char* __restrict__ wsX,
    float* __restrict__ out) {
  const int lane = threadIdx.x & 31;
  const int wave = threadIdx.x >> 5;
  const int wm = wave & 3;                     // 4 waves along M
  const int wn = wave >> 2;                    // 2 waves along N
  const int mtb = blockIdx.y * 8 + wm * 2;     // base 16-row tile (2 per wave)
  const int ntb = blockIdx.x * 8 + wn * 4;     // base 16-col tile (4 per wave)

  v8f accR[2][4] = {};
  v8f accI[2][4] = {};

  const unsigned char* aBase[2];
  #pragma unroll
  for (int im = 0; im < 2; ++im)
    aBase[im] = wsX + ((size_t)(mtb + im) * KT_TILES) * X_TILE_BYTES + (size_t)lane * 32;

  const unsigned char* bBase[4];
  #pragma unroll
  for (int jn = 0; jn < 4; ++jn)
    bBase[jn] = wsW + ((size_t)(ntb + jn) * KT_TILES) * W_TILE_BYTES + (size_t)lane * 32;

  for (int kt = 0; kt < KT_TILES; ++kt) {
    v16h a[2];
    #pragma unroll
    for (int im = 0; im < 2; ++im)
      a[im] = *(const v16h*)(aBase[im] + (size_t)kt * X_TILE_BYTES);

    v16h br[4], bi[4];
    #pragma unroll
    for (int jn = 0; jn < 4; ++jn) {
      br[jn] = *(const v16h*)(bBase[jn] + (size_t)kt * W_TILE_BYTES);
      bi[jn] = *(const v16h*)(bBase[jn] + (size_t)kt * W_TILE_BYTES + 1024);
    }

    #pragma unroll
    for (int im = 0; im < 2; ++im) {
      #pragma unroll
      for (int jn = 0; jn < 4; ++jn) {
        accR[im][jn] = __builtin_amdgcn_wmma_f32_16x16x32_f16(
            false, a[im], false, br[jn], (short)0, accR[im][jn], false, false);
        accI[im][jn] = __builtin_amdgcn_wmma_f32_16x16x32_f16(
            false, a[im], false, bi[jn], (short)0, accI[im][jn], false, false);
      }
    }
  }

  // Epilogue: C/D layout -> VGPR r holds M = r (lanes 0-15) / M = 8+r (lanes
  // 16-31), N = lane & 15. Output is interleaved complex64: coalesced b64.
  const int mhalf = (lane & 16) ? 8 : 0;
  const int ncol  = lane & 15;
  #pragma unroll
  for (int im = 0; im < 2; ++im) {
    #pragma unroll
    for (int jn = 0; jn < 4; ++jn) {
      const int ng = (ntb + jn) * 16 + ncol;
      #pragma unroll
      for (int r = 0; r < 8; ++r) {
        const int mg = (mtb + im) * 16 + mhalf + r;
        float2 o;
        o.x = accR[im][jn][r];
        o.y = accI[im][jn][r];
        *(float2*)(out + ((size_t)mg * N_FFT + ng) * 2) = o;
      }
    }
  }
}

extern "C" void kernel_launch(void* const* d_in, const int* in_sizes, int n_in,
                              void* d_out, int out_size, void* d_ws, size_t ws_size,
                              hipStream_t stream) {
  const float* x = (const float*)d_in[0];
  float* out = (float*)d_out;

  unsigned char* ws  = (unsigned char*)d_ws;
  unsigned char* wsW = ws;                                              // 64 MB
  unsigned char* wsX = ws + (size_t)NT_TILES * KT_TILES * W_TILE_BYTES; // +32 MB

  const int M  = in_sizes[0] / N_FFT;   // 4*1024 = 4096 rows
  const int MT = M / 16;                // 256 M-tiles

  // 1) Build Wr/Wi f16 B-fragments (one wave per 32x16 tile, 8 tiles/block).
  gen_w_kernel<<<(NT_TILES * KT_TILES) / 8, 256, 0, stream>>>(wsW);

  // 2) Convert x to f16 A-fragments.
  gen_x_kernel<<<(MT * KT_TILES) / 8, 256, 0, stream>>>(x, wsX);

  // 3) Fused real+imag causal-DFT GEMM: 32 x 32 blocks of 128x128 outputs.
  dim3 grid(N_FFT / 128, M / 128);
  causal_dft_wmma_kernel<<<grid, 256, 0, stream>>>(wsW, wsX, out);
}
